// GroupedQueryAttention_37864431681667
// MI455X (gfx1250) — compile-verified, matrix operand order checked by reading, not run on hardware
//
#include <hip/hip_runtime.h>

#define B_   2
#define S_   2048
#define H_   2048
#define NH_  16
#define NKV_ 4
#define HD_  128
#define KVD_ (NKV_ * HD_)   /* 512 */
#define MROWS_ (B_ * S_)    /* 4096 */

typedef __attribute__((ext_vector_type(16))) __bf16 v16bf;
typedef __attribute__((ext_vector_type(8)))  __bf16 v8bf;
typedef __attribute__((ext_vector_type(8)))  float  v8f;

__device__ __forceinline__ v8f wmma_bf16(v16bf a, v16bf b, v8f c) {
  return __builtin_amdgcn_wmma_f32_16x16x32_bf16(
      /*neg_a=*/false, a, /*neg_b=*/false, b,
      /*c_mod=*/(short)0, c, /*reuse_a=*/false, /*reuse_b=*/false);
}

// Load one 16x32 bf16 WMMA operand (A-pattern; B-operand uses the same pattern
// when the matrix is stored transposed [N,K] row-major).
// lane m (0..15): row m, K chunks {0..7, 16..23}; lane m+16: row m, K {8..15, 24..31}.
__device__ __forceinline__ v16bf load_op16x32(const __bf16* base, long ld,
                                              long row, long k0, int lane) {
  const __bf16* p = base + row * ld + k0 + (long)((lane >> 4) << 3);
  v8bf lo = *(const v8bf*)(p);
  v8bf hi = *(const v8bf*)(p + 16);
  v16bf r;
#pragma unroll
  for (int i = 0; i < 8; ++i) { r[i] = lo[i]; r[i + 8] = hi[i]; }
  return r;
}

// ---------------- elementwise fp32 -> bf16 ----------------
__global__ __launch_bounds__(256) void f32_to_bf16_kernel(const float* __restrict__ in,
                                                          __bf16* __restrict__ out, long n) {
  long i = (long)blockIdx.x * blockDim.x + threadIdx.x;
  if (i < n) out[i] = (__bf16)in[i];
}

// ---------------- W[K,N] fp32 -> Wt[N,K] bf16, 32x32 LDS-tiled ----------------
// grid: (K/32, N/32), block 256 (8 rows of 32).
__global__ __launch_bounds__(256) void transpose_to_bf16_kernel(const float* __restrict__ W,
                                                                __bf16* __restrict__ Wt,
                                                                int K, int N) {
  __shared__ float tile[32][33];
  const int k0 = blockIdx.x * 32, n0 = blockIdx.y * 32;
  const int tx = threadIdx.x & 31, ty = threadIdx.x >> 5;  // ty in 0..7
#pragma unroll
  for (int i = 0; i < 32; i += 8)
    tile[ty + i][tx] = W[(long)(k0 + ty + i) * N + (n0 + tx)];
  __syncthreads();
#pragma unroll
  for (int i = 0; i < 32; i += 8)
    Wt[(long)(n0 + ty + i) * K + (k0 + tx)] = (__bf16)tile[tx][ty + i];
}

// ---------------- generic WMMA GEMM: C[M,N] = A[M,K] @ Bt[N,K]^T + bias -----
// Block tile 256(M) x 128(N); 8 waves arranged 4(M) x 2(N); wave tile 64x64
// = 4x4 WMMA accumulators (A and B operands each reused 4x).
// MODE 0: bf16 out row-major   MODE 1: bf16 out, V-transposed store Vt[b,kvh,HD,S]
// MODE 2: fp32 out row-major
template <int NN, int KK, int MODE>
__global__ __launch_bounds__(256) void gemm_bf16_wmma(const __bf16* __restrict__ A,
                                                      const __bf16* __restrict__ Bt,
                                                      const float* __restrict__ bias,
                                                      void* __restrict__ Cout,
                                                      float scale) {
  const int lane = threadIdx.x & 31;
  const int wid  = threadIdx.x >> 5;
  const int wm = wid >> 1;       // 4 waves along M (64 rows each)
  const int wn = wid & 1;        // 2 waves along N (64 cols each)
  const long m0 = (long)blockIdx.y * 256 + wm * 64;
  const long n0 = (long)blockIdx.x * 128 + wn * 64;
  const int cl = lane & 15;
  const int rh = (lane >> 4) << 3;

  v8f acc[4][4];
#pragma unroll
  for (int mt = 0; mt < 4; ++mt)
#pragma unroll
    for (int nt = 0; nt < 4; ++nt) acc[mt][nt] = {};

  for (int kk = 0; kk < KK; kk += 32) {
    v16bf a[4];
#pragma unroll
    for (int mt = 0; mt < 4; ++mt)
      a[mt] = load_op16x32(A, KK, m0 + mt * 16 + cl, kk, lane);
#pragma unroll
    for (int nt = 0; nt < 4; ++nt) {
      v16bf b = load_op16x32(Bt, KK, n0 + nt * 16 + cl, kk, lane);
#pragma unroll
      for (int mt = 0; mt < 4; ++mt)
        acc[mt][nt] = wmma_bf16(a[mt], b, acc[mt][nt]);
    }
  }

#pragma unroll
  for (int mt = 0; mt < 4; ++mt) {
#pragma unroll
    for (int nt = 0; nt < 4; ++nt) {
      const long col = n0 + nt * 16 + cl;
      const float bv = bias[col];
      const long r0 = m0 + mt * 16 + rh;
#pragma unroll
      for (int i = 0; i < 8; ++i) {
        const float v = (acc[mt][nt][i] + bv) * scale;
        const long rr = r0 + i;
        if constexpr (MODE == 0) {
          ((__bf16*)Cout)[rr * NN + col] = (__bf16)v;
        } else if constexpr (MODE == 2) {
          ((float*)Cout)[rr * NN + col] = v;
        } else {  // V transposed store: rr -> (b,s), col -> (kvh,d); all pow2
          const long bb = rr / S_, ss = rr % S_;
          const long kvh = col / HD_, d = col % HD_;
          ((__bf16*)Cout)[(((bb * NKV_) + kvh) * HD_ + d) * S_ + ss] = (__bf16)v;
        }
      }
    }
  }
}

// ---------------- flash GQA attention ----------------
// grid (S/64, NH, B), block 128 (4 waves). Each wave: 16 queries x full HD.
__global__ __launch_bounds__(128) void gqa_flash_wmma(const __bf16* __restrict__ Qb,
                                                      const __bf16* __restrict__ Kb,
                                                      const __bf16* __restrict__ Vt,
                                                      const int* __restrict__ amask,
                                                      __bf16* __restrict__ ctxb) {
  __shared__ alignas(16) __bf16 psh[4][16 * 32];

  const int lane = threadIdx.x & 31;
  const int w    = threadIdx.x >> 5;
  const int h = blockIdx.y, b = blockIdx.z;
  const int kvh = h / (NH_ / NKV_);
  const long qbase = (long)blockIdx.x * 64 + w * 16;
  const int cl = lane & 15;
  const int ch = lane >> 4;

  const __bf16* Qp = Qb + (long)b * S_ * H_ + (long)h * HD_;          // row=s, ld=H
  const __bf16* Kp = Kb + (long)b * S_ * KVD_ + (long)kvh * HD_;      // row=key, ld=KVD
  const __bf16* Vp = Vt + ((long)(b * NKV_ + kvh)) * HD_ * S_;        // row=d, ld=S
  const int* maskrow = amask + (long)b * S_;

  v16bf q[4];
#pragma unroll
  for (int i = 0; i < 4; ++i)
    q[i] = load_op16x32(Qp, H_, qbase + cl, i * 32, lane);

  v8f ctx[8];
#pragma unroll
  for (int dt = 0; dt < 8; ++dt) ctx[dt] = {};
  float m[8], l[8];
#pragma unroll
  for (int r = 0; r < 8; ++r) { m[r] = -3.0e38f; l[r] = 0.f; }

  for (int kb = 0; kb < S_; kb += 32) {
    // ---- scores: 16 queries x 32 keys, contraction over HD=128 ----
    v8f s0 = {}, s1 = {};
#pragma unroll
    for (int i = 0; i < 4; ++i) {
      v16bf k0 = load_op16x32(Kp, KVD_, kb + cl, i * 32, lane);
      v16bf k1 = load_op16x32(Kp, KVD_, kb + 16 + cl, i * 32, lane);
      s0 = wmma_bf16(q[i], k0, s0);
      s1 = wmma_bf16(q[i], k1, s1);
    }
    // attention mask (column = key index held by this lane)
    const bool z0 = (maskrow[kb + cl] == 0);
    const bool z1 = (maskrow[kb + 16 + cl] == 0);
#pragma unroll
    for (int i = 0; i < 8; ++i) {
      s0[i] = z0 ? -3.0e38f : s0[i];
      s1[i] = z1 ? -3.0e38f : s1[i];
    }
    // ---- online softmax over the 32-key tile (row-wise across 16 lanes) ----
    float fscale[8];
#pragma unroll
    for (int r = 0; r < 8; ++r) {
      float mx = fmaxf(s0[r], s1[r]);
#pragma unroll
      for (int off = 1; off < 16; off <<= 1) mx = fmaxf(mx, __shfl_xor(mx, off, 32));
      float mn = fmaxf(m[r], mx);
      float f = __expf(m[r] - mn);
      float p0 = __expf(s0[r] - mn);
      float p1 = __expf(s1[r] - mn);
      s0[r] = p0; s1[r] = p1;
      float rs = p0 + p1;
#pragma unroll
      for (int off = 1; off < 16; off <<= 1) rs += __shfl_xor(rs, off, 32);
      l[r] = l[r] * f + rs;
      m[r] = mn;
      fscale[r] = f;
    }
#pragma unroll
    for (int dt = 0; dt < 8; ++dt)
#pragma unroll
      for (int r = 0; r < 8; ++r) ctx[dt][r] *= fscale[r];

    // ---- C-layout -> A-layout transpose of P through LDS ----
    __bf16* P = psh[w];
#pragma unroll
    for (int r = 0; r < 8; ++r) {
      P[(ch * 8 + r) * 32 + cl]      = (__bf16)s0[r];
      P[(ch * 8 + r) * 32 + 16 + cl] = (__bf16)s1[r];
    }
    asm volatile("s_wait_dscnt 0x0" ::: "memory");
    v16bf pa = load_op16x32(P, 32, cl, 0, lane);

    // ---- ctx += P @ V  (B-operand = rows of Vt) ----
#pragma unroll
    for (int dt = 0; dt < 8; ++dt) {
      v16bf vb = load_op16x32(Vp, S_, dt * 16 + cl, kb, lane);
      ctx[dt] = wmma_bf16(pa, vb, ctx[dt]);
    }
  }

  // ---- normalize and store ctx (head-interleaved [B,S,H]) ----
  float inv[8];
#pragma unroll
  for (int r = 0; r < 8; ++r) inv[r] = (l[r] > 0.f) ? (1.f / l[r]) : 0.f;
  __bf16* outp = ctxb + ((long)b * S_ + qbase) * H_ + (long)h * HD_;
#pragma unroll
  for (int dt = 0; dt < 8; ++dt)
#pragma unroll
    for (int r = 0; r < 8; ++r)
      outp[(long)(ch * 8 + r) * H_ + dt * 16 + cl] = (__bf16)(ctx[dt][r] * inv[r]);
}

extern "C" void kernel_launch(void* const* d_in, const int* in_sizes, int n_in,
                              void* d_out, int out_size, void* d_ws, size_t ws_size,
                              hipStream_t stream) {
  (void)in_sizes; (void)n_in; (void)out_size; (void)ws_size;
  const float* x  = (const float*)d_in[0];
  const int*   am = (const int*)d_in[1];
  const float* Wq = (const float*)d_in[2];
  const float* bq = (const float*)d_in[3];
  const float* Wk = (const float*)d_in[4];
  const float* bk = (const float*)d_in[5];
  const float* Wv = (const float*)d_in[6];
  const float* bv = (const float*)d_in[7];
  const float* Wo = (const float*)d_in[8];
  const float* bo = (const float*)d_in[9];

  char* ws = (char*)d_ws;
  const long SZ_X   = (long)MROWS_ * H_ * 2;    // 16.8 MB bf16
  const long SZ_WQ  = (long)H_ * H_ * 2;        // 8.4 MB
  const long SZ_WKV = (long)H_ * KVD_ * 2;      // 2.1 MB
  const long SZ_QB  = (long)MROWS_ * H_ * 2;
  const long SZ_KB  = (long)MROWS_ * KVD_ * 2;

  __bf16* xb   = (__bf16*)(ws);                 long off = SZ_X;
  __bf16* WqT  = (__bf16*)(ws + off);           off += SZ_WQ;
  __bf16* WkT  = (__bf16*)(ws + off);           off += SZ_WKV;
  __bf16* WvT  = (__bf16*)(ws + off);           off += SZ_WKV;
  __bf16* WoT  = (__bf16*)(ws + off);           off += SZ_WQ;
  __bf16* Qb   = (__bf16*)(ws + off);           off += SZ_QB;
  __bf16* Kb   = (__bf16*)(ws + off);           off += SZ_KB;
  __bf16* Vtb  = (__bf16*)(ws + off);           off += SZ_KB;
  __bf16* ctxb = (__bf16*)(ws + off);

  const float qscale = 0.08838834764831845f;    // 1/sqrt(128)

  // 1) conversions
  {
    long n = (long)MROWS_ * H_;
    f32_to_bf16_kernel<<<(unsigned)(n / 256), 256, 0, stream>>>(x, xb, n);
  }
  transpose_to_bf16_kernel<<<dim3(H_ / 32, H_ / 32), 256, 0, stream>>>(Wq, WqT, H_, H_);
  transpose_to_bf16_kernel<<<dim3(H_ / 32, KVD_ / 32), 256, 0, stream>>>(Wk, WkT, H_, KVD_);
  transpose_to_bf16_kernel<<<dim3(H_ / 32, KVD_ / 32), 256, 0, stream>>>(Wv, WvT, H_, KVD_);
  transpose_to_bf16_kernel<<<dim3(H_ / 32, H_ / 32), 256, 0, stream>>>(Wo, WoT, H_, H_);

  // 2) projections
  gemm_bf16_wmma<H_, H_, 0><<<dim3(H_ / 128, MROWS_ / 256), 256, 0, stream>>>(
      xb, WqT, bq, (void*)Qb, qscale);
  gemm_bf16_wmma<KVD_, H_, 0><<<dim3(KVD_ / 128, MROWS_ / 256), 256, 0, stream>>>(
      xb, WkT, bk, (void*)Kb, 1.0f);
  gemm_bf16_wmma<KVD_, H_, 1><<<dim3(KVD_ / 128, MROWS_ / 256), 256, 0, stream>>>(
      xb, WvT, bv, (void*)Vtb, 1.0f);

  // 3) flash attention
  gqa_flash_wmma<<<dim3(S_ / 64, NH_, B_), 128, 0, stream>>>(Qb, Kb, Vtb, am, ctxb);

  // 4) output projection -> fp32 d_out
  gemm_bf16_wmma<H_, H_, 2><<<dim3(H_ / 128, MROWS_ / 256), 256, 0, stream>>>(
      ctxb, WoT, bo, d_out, 1.0f);
}
